// MenuLoss_7713761264358
// MI455X (gfx1250) — matrix operand order repaired and checked by reading, not compiled, not run on hardware
//
#include <hip/hip_runtime.h>
#include <hip/hip_bf16.h>
#include <math.h>

typedef __attribute__((ext_vector_type(16))) _Float16 v16h;
typedef __attribute__((ext_vector_type(8)))  float    v8f;

#define N_FOODS   223
#define NCOLS     19
#define EPB       168      // elements per batch row: 7*3*8
#define NTILES    11       // ceil(168/16)
#define EPAD      176      // NTILES*16
#define KPAD      224      // 7 chunks of K=32
#define ACC_SLOTS 53
#define WS_STRIDE 56

__device__ __forceinline__ float sigmoidf_(float x) {
  return 1.0f / (1.0f + __expf(-x));
}

__device__ __forceinline__ float huber_(float x) {
  float d = fabsf(x);
  return (d < 1.0f) ? 0.5f * d * d : (d - 0.5f);
}

// ---------------------------------------------------------------------------
// Phase 1: one block per batch row b. Computes W@data for pred and gold paths
// with V_WMMA_F32_16X16X32_F16 and reduces everything to 53 scalars per b.
// ---------------------------------------------------------------------------
__global__ __launch_bounds__(128)
void menu_phase1(const float* __restrict__ y_pred,
                 const float* __restrict__ y,
                 const float* __restrict__ data,
                 float* __restrict__ ws)
{
  // B matrix staged column-major [col][K] in f16, zero padded (cols 19..31, K=223)
  __shared__ _Float16 sB[32 * KPAD];
  __shared__ float sRP[EPAD], sTID[EPAD], sAP[EPAD], sAT[EPAD];
  __shared__ float sAcc[ACC_SLOTS + 3];

  const int tid = threadIdx.x;
  const int b   = blockIdx.x;

  for (int i = tid; i < ACC_SLOTS + 3; i += blockDim.x) sAcc[i] = 0.0f;

  for (int i = tid; i < 32 * KPAD; i += blockDim.x) {
    int col = i / KPAD, K = i % KPAD;
    float v = (col < NCOLS && K < N_FOODS) ? data[K * NCOLS + col] : 0.0f;
    sB[i] = (_Float16)v;
  }
  __syncthreads();

  // Stage per-element values; fold zeros_penalty / id_range on the way.
  const float* yp = y_pred + (size_t)b * EPB * 2;
  const float* yt = y      + (size_t)b * EPB * 2;
  float zacc = 0.0f, iacc = 0.0f;
  for (int e = tid; e < EPAD; e += blockDim.x) {
    if (e < EPB) {
      float pid = yp[e * 2 + 0], pam = yp[e * 2 + 1];
      float tiv = yt[e * 2 + 0], tam = yt[e * 2 + 1];
      float r  = rintf(pid);                               // round-to-even like jnp.round
      float rp = r * sigmoidf_(50.0f * (222.5f - r));      // _mask
      sRP[e] = rp; sTID[e] = tiv; sAP[e] = pam; sAT[e] = tam;
      zacc += 2.0f * (1.0f - tanhf(2.0f * pid)) + (1.0f - tanhf(2.0f * pam));
      iacc += fmaxf(pid - 222.0f, 0.0f);
    } else {
      sRP[e] = -1.0e4f; sTID[e] = -1.0e4f; sAP[e] = 0.0f; sAT[e] = 0.0f;
    }
  }
  atomicAdd(&sAcc[51], zacc);
  atomicAdd(&sAcc[52], iacc);
  __syncthreads();

  const int lane = tid & 31;
  const int wave = tid >> 5;
  const int col  = lane & 15;   // C/D: lane owns this column (0..15 / 16..31)
  const int half = lane >> 4;
  const bool colLT7 = (col < 7);

  // Every lane carries all accumulators; only the correct ones are stored at
  // the end, so wrong-column lanes can accumulate garbage-free zero-cost junk.
  float accNutP = 0, accNutT = 0, accBinP = 0, accBinT = 0, accB2P = 0, accB2T = 0;
  float mp0 = 0, mp1 = 0, mp2 = 0, mt0 = 0, mt1 = 0, mt2 = 0;
  float d0 = 0, d1 = 0, d2 = 0, d3 = 0, d4 = 0, d5 = 0, d6 = 0;

  for (int t = wave; t < NTILES; t += 4) {
    // A-layout: this lane supplies row M = lane%16 of the tile.
    float xp = sRP[t * 16 + col];
    float xt = sTID[t * 16 + col];
    v8f cP0 = {}, cP1 = {}, cT0 = {}, cT1 = {};

    for (int q = 0; q < 7; ++q) {
      const int kb = q * 32;
      v16h aP, aT, b0, b1;
#pragma unroll
      for (int j = 0; j < 16; ++j) {
        // 16-bit A 16x32: lanes 0-15 K={0..7,16..23}, lanes 16-31 K={8..15,24..31}
        int ka = kb + ((j < 8) ? (half * 8 + j) : (16 + half * 8 + (j - 8)));
        float dp = xp - (float)ka;
        float dt = xt - (float)ka;
        aP[j] = (_Float16)__expf(-100.0f * dp * dp);
        aT[j] = (_Float16)__expf(-100.0f * dt * dt);
        // 16-bit B 32x16: lanes 0-15 K={0..7,16..23}, lanes 16-31 K={8..15,24..31}
        int kbv = ka;
        b0[j] = sB[col * KPAD + kbv];
        b1[j] = sB[(16 + col) * KPAD + kbv];
      }
      cP0 = __builtin_amdgcn_wmma_f32_16x16x32_f16(false, aP, false, b0, (short)0, cP0, false, false);
      cP1 = __builtin_amdgcn_wmma_f32_16x16x32_f16(false, aP, false, b1, (short)0, cP1, false, false);
      cT0 = __builtin_amdgcn_wmma_f32_16x16x32_f16(false, aT, false, b0, (short)0, cT0, false, false);
      cT1 = __builtin_amdgcn_wmma_f32_16x16x32_f16(false, aT, false, b1, (short)0, cT1, false, false);
    }

    // C/D layout: VGPR v holds row M = v + 8*half, column = lane%16.
    // Branchless accumulation: padded elements (e >= 168) have zero weights and
    // zero amounts, so every term is naturally zero except the (1 - val)
    // pref columns, which get one select against vmask.
    float cpArr[8], ctArr[8];
#pragma unroll
    for (int v = 0; v < 8; ++v) {
      int e = t * 16 + v + 8 * half;
      float vmask = (e < EPB) ? 1.0f : 0.0f;
      float ap = sAP[e], at = sAT[e];
      float vp0 = cP0[v], vt0 = cT0[v];
      float cp = vp0 * ap, ct = vt0 * at;
      cpArr[v] = cp; ctArr[v] = ct;
      accNutP += cp;                 // meaningful only for col<5 lanes
      accNutT += ct;
      // col 5,6: valid*(1-val) == vmask - val (val==0 when invalid); col>=7: val
      accBinP += colLT7 ? (vmask - vp0) : vp0;
      accBinT += colLT7 ? (vmask - vt0) : vt0;
      accB2P  += cP1[v];             // meaningful only for col<3 lanes
      accB2T  += cT1[v];
    }

    if (col == 0) {                  // CAL lanes also feed meals + per-day cals
#pragma unroll
      for (int v = 0; v < 8; ++v) {
        int e = t * 16 + v + 8 * half;
        int m = (e >> 3) % 3;
        int d = e / 24;
        float cp = cpArr[v], ct = ctArr[v];
        mp0 += (m == 0) ? cp : 0.0f; mp1 += (m == 1) ? cp : 0.0f; mp2 += (m == 2) ? cp : 0.0f;
        mt0 += (m == 0) ? ct : 0.0f; mt1 += (m == 1) ? ct : 0.0f; mt2 += (m == 2) ? ct : 0.0f;
        d0 += (d == 0) ? cp : 0.0f; d1 += (d == 1) ? cp : 0.0f; d2 += (d == 2) ? cp : 0.0f;
        d3 += (d == 3) ? cp : 0.0f; d4 += (d == 4) ? cp : 0.0f; d5 += (d == 5) ? cp : 0.0f;
        d6 += (d == 6) ? cp : 0.0f;
      }
    }
  }

  // Slot map: [0..4] nutP, [5..9] nutT, [10..23] binP(c=5..18),
  // [24..37] binT, [38..40] mealP, [41..43] mealT, [44..50] dayP, [51] z, [52] idr
  if (col < 5) {
    atomicAdd(&sAcc[col],     accNutP);
    atomicAdd(&sAcc[5 + col], accNutT);
  } else {
    atomicAdd(&sAcc[10 + (col - 5)], accBinP);
    atomicAdd(&sAcc[24 + (col - 5)], accBinT);
  }
  if (col < 3) {
    atomicAdd(&sAcc[21 + col], accB2P);   // cols 16..18 -> idx 11..13
    atomicAdd(&sAcc[35 + col], accB2T);
  }
  if (col == 0) {
    atomicAdd(&sAcc[38], mp0); atomicAdd(&sAcc[39], mp1); atomicAdd(&sAcc[40], mp2);
    atomicAdd(&sAcc[41], mt0); atomicAdd(&sAcc[42], mt1); atomicAdd(&sAcc[43], mt2);
    atomicAdd(&sAcc[44], d0);  atomicAdd(&sAcc[45], d1);  atomicAdd(&sAcc[46], d2);
    atomicAdd(&sAcc[47], d3);  atomicAdd(&sAcc[48], d4);  atomicAdd(&sAcc[49], d5);
    atomicAdd(&sAcc[50], d6);
  }
  __syncthreads();

  if (tid < ACC_SLOTS) ws[(size_t)b * WS_STRIDE + tid] = sAcc[tid];
}

// ---------------------------------------------------------------------------
// Phase 2: thread b finalizes per-batch terms; block-reduce to scalar loss.
// ---------------------------------------------------------------------------
__global__ __launch_bounds__(512)
void menu_phase2(const float* __restrict__ ws, float* __restrict__ out)
{
  __shared__ float red[512];
  const int b = threadIdx.x;
  const float* w = ws + (size_t)b * WS_STRIDE;
  float acc = 0.0f;

  // nutrition_diff: stored sum(val*amt); /100 then /7 => /700
  for (int c = 0; c < 5; ++c)
    acc += huber_((w[c] - w[5 + c]) * (1.0f / 700.0f)) * 0.01f;

  // preferences (c=5,6) + allergens (c=7..13): 100 * exp(-10*gold) * pred^2
  for (int i = 0; i < 9; ++i) {
    float g = w[24 + i], p = w[10 + i];
    acc += 100.0f * __expf(-10.0f * g) * p * p;
  }

  // ingredients (c=14..18): smooth_l1 / 100
  for (int i = 9; i < 14; ++i)
    acc += huber_(w[10 + i] - w[24 + i]) * 0.01f;

  // meals_diff
  for (int i = 0; i < 3; ++i)
    acc += huber_((w[38 + i] - w[41 + i]) * (1.0f / 700.0f)) * 0.01f;

  // per-day calorie variance (pred path): day value = sum(val*amt)/100
  float mu = 0.0f;
  for (int d = 0; d < 7; ++d) mu += w[44 + d] * 0.01f;
  mu *= (1.0f / 7.0f);
  float var = 0.0f;
  for (int d = 0; d < 7; ++d) {
    float x = w[44 + d] * 0.01f - mu;
    var += x * x;
  }
  acc += var * (1.0f / 7.0f);

  // zeros_penalty + id_range_penalty
  acc += 3000.0f * w[51] + w[52];

  red[b] = acc;
  __syncthreads();
  for (int s = 256; s > 0; s >>= 1) {
    if (b < s) red[b] += red[b + s];
    __syncthreads();
  }
  if (b == 0) out[0] = red[0] * (1.0f / 512.0f);
}

extern "C" void kernel_launch(void* const* d_in, const int* in_sizes, int n_in,
                              void* d_out, int out_size, void* d_ws, size_t ws_size,
                              hipStream_t stream) {
  const float* y_pred = (const float*)d_in[0];   // (512,7,3,8,2) f32
  const float* y      = (const float*)d_in[1];   // (512,7,3,8,2) f32
  const float* data   = (const float*)d_in[2];   // (223,19) f32
  float* ws = (float*)d_ws;                      // 512*56 floats used
  menu_phase1<<<512, 128, 0, stream>>>(y_pred, y, data, ws);
  menu_phase2<<<1, 512, 0, stream>>>(ws, (float*)d_out);
}
